// InvariantReadout_71803263254730
// MI455X (gfx1250) — matrix operand order repaired and checked by reading, not compiled, not run on hardware
//
#include <hip/hip_runtime.h>
#include <hip/hip_bf16.h>
#include <math.h>

typedef __bf16 bf16_t;
typedef __attribute__((ext_vector_type(16))) __bf16 v16bf;
typedef __attribute__((ext_vector_type(8)))  __bf16 v8bf;
typedef __attribute__((ext_vector_type(4)))  __bf16 v4bf;
typedef __attribute__((ext_vector_type(8)))  float  v8f;

// ---------- helpers ----------
__device__ __forceinline__ unsigned fmap(float f) {
    int i = __float_as_int(f);
    return (i >= 0) ? ((unsigned)i ^ 0x80000000u) : ~(unsigned)i;
}
__device__ __forceinline__ float funmap(unsigned u) {
    int i = (u & 0x80000000u) ? (int)(u ^ 0x80000000u) : (int)~u;
    return __int_as_float(i);
}
// fast SiLU: x * sigmoid(x) using v_rcp_f32 instead of IEEE divide expansion
__device__ __forceinline__ float fast_silu(float v) {
    return v * __builtin_amdgcn_rcpf(1.0f + __expf(-v));
}

// One 16x256 = (16x256 strip) @ (256x256 W) layer with bias + SiLU.
// src/dst are per-wave LDS strips (16 rows x 256 cols, bf16, row-major).
// wb is the pre-packed B-fragment weight layout: frag(nt,kt) at (nt*8+kt)*512
// halves; within a frag: lane*16 contiguous bf16 (lane= N%16 + 16*(K-half)).
__device__ __forceinline__ void layer_silu(const bf16_t* src, bf16_t* dst,
                                           const bf16_t* __restrict__ wb,
                                           const float* sbias, int lane)
{
    union Frag { v16bf v; v8bf h[2]; };
    Frag A[8], B;
    const int mrow  = lane & 15;
    const int khalf = (lane >> 4) * 8;
#pragma unroll
    for (int kt = 0; kt < 8; ++kt) {
        int off = mrow * 256 + kt * 32 + khalf;
        A[kt].h[0] = *(const v8bf*)(src + off);
        A[kt].h[1] = *(const v8bf*)(src + off + 16);
    }
    const int ncol  = lane & 15;
    const int mbase = (lane >> 4) * 8;
    for (int nt = 0; nt < 16; ++nt) {
        float bv = sbias[nt * 16 + ncol];
        v8f acc;
#pragma unroll
        for (int j = 0; j < 8; ++j) acc[j] = bv;
#pragma unroll
        for (int kt = 0; kt < 8; ++kt) {
            const bf16_t* wp = wb + ((nt * 8 + kt) * 512 + lane * 16);
            B.h[0] = *(const v8bf*)(wp);
            B.h[1] = *(const v8bf*)(wp + 8);
            acc = __builtin_amdgcn_wmma_f32_16x16x32_bf16(
                false, A[kt].v, false, B.v, (short)0, acc, false, false);
        }
#pragma unroll
        for (int j = 0; j < 8; ++j) {
            dst[(mbase + j) * 256 + nt * 16 + ncol] = (bf16_t)fast_silu(acc[j]);
        }
    }
}

// ---------- kernel 0: repack weights f32 -> bf16 B-fragment layout ----------
__global__ __launch_bounds__(256) void convert_weights(
    const float* __restrict__ W1, const float* __restrict__ W2,
    const float* __restrict__ W3, const float* __restrict__ WA,
    bf16_t* __restrict__ B1, bf16_t* __restrict__ B2,
    bf16_t* __restrict__ B3, bf16_t* __restrict__ BA)
{
    int tid = blockIdx.x * blockDim.x + threadIdx.x;   // 0 .. 4*65536-1
    int mat = tid >> 16;
    int e   = tid & 65535;
    int f   = e >> 9;            // fragment index 0..127 (nt*8+kt)
    int r   = e & 511;
    int lane = r >> 4;
    int h    = r & 15;
    int nt = f >> 3, kt = f & 7;
    int k = kt * 32 + ((lane >> 4) << 3) + (h & 7) + ((h >> 3) << 4);  // K index
    int n = nt * 16 + (lane & 15);                     // N index
    const float* src = (mat == 0) ? W1 : (mat == 1) ? W2 : (mat == 2) ? W3 : WA;
    bf16_t*      dst = (mat == 0) ? B1 : (mat == 1) ? B2 : (mat == 2) ? B3 : BA;
    dst[e] = (bf16_t)src[k * 256 + n];
}

// ---------- kernel 1: zero out / init segment reductions ----------
__global__ __launch_bounds__(256) void init_kernel(float* out, unsigned* m_u,
                                                   float* denom, int outN, int G)
{
    int i = blockIdx.x * blockDim.x + threadIdx.x;
    if (i < outN) out[i] = 0.f;
    if (i < G) { m_u[i] = 0u; denom[i] = 0.f; }
}

// ---------- kernel 2: attention branch -> logits + segment max ----------
__global__ __launch_bounds__(128) void attn_kernel(
    const float* __restrict__ X, const int* __restrict__ batch,
    const bf16_t* __restrict__ WbA, const float* __restrict__ ab1,
    const float* __restrict__ aW2, const float* __restrict__ ab2,
    float* __restrict__ logits, unsigned* __restrict__ m_u, int Ntot)
{
    __shared__ __align__(16) bf16_t strip[4 * 16 * 256];
    __shared__ float s_aw2[256];
    __shared__ float s_ab1[256];
    const int wv   = threadIdx.x >> 5;
    const int lane = threadIdx.x & 31;
    const int node0 = (blockIdx.x * 4 + wv) * 16;
    bf16_t* s = strip + wv * 4096;

    for (int i = threadIdx.x; i < 256; i += 128) { s_aw2[i] = aW2[i]; s_ab1[i] = ab1[i]; }

    // stage 16x256 node-feature strip as bf16
    const float4* X4 = (const float4*)X;
    for (int i = 0; i < 32; ++i) {
        int idx4 = i * 32 + lane;
        int m = idx4 >> 6, c4 = idx4 & 63;
        int node = node0 + m;
        float4 f;
        if (node < Ntot) f = X4[(size_t)node * 64 + c4];
        else             f = make_float4(0.f, 0.f, 0.f, 0.f);
        v4bf b; b[0] = (bf16_t)f.x; b[1] = (bf16_t)f.y; b[2] = (bf16_t)f.z; b[3] = (bf16_t)f.w;
        *(v4bf*)(s + m * 256 + c4 * 4) = b;
    }
    __syncthreads();

    union Frag { v16bf v; v8bf h[2]; };
    Frag A[8], B;
    const int mrow = lane & 15, khalf = (lane >> 4) * 8;
#pragma unroll
    for (int kt = 0; kt < 8; ++kt) {
        int off = mrow * 256 + kt * 32 + khalf;
        A[kt].h[0] = *(const v8bf*)(s + off);
        A[kt].h[1] = *(const v8bf*)(s + off + 16);
    }
    const int ncol = lane & 15, mbase = (lane >> 4) * 8;
    float part[8];
#pragma unroll
    for (int j = 0; j < 8; ++j) part[j] = 0.f;

    for (int nt = 0; nt < 16; ++nt) {
        float bv = s_ab1[nt * 16 + ncol];
        v8f acc;
#pragma unroll
        for (int j = 0; j < 8; ++j) acc[j] = bv;
#pragma unroll
        for (int kt = 0; kt < 8; ++kt) {
            const bf16_t* wp = WbA + ((nt * 8 + kt) * 512 + lane * 16);
            B.h[0] = *(const v8bf*)(wp);
            B.h[1] = *(const v8bf*)(wp + 8);
            acc = __builtin_amdgcn_wmma_f32_16x16x32_bf16(
                false, A[kt].v, false, B.v, (short)0, acc, false, false);
        }
        float wn = s_aw2[nt * 16 + ncol];
#pragma unroll
        for (int j = 0; j < 8; ++j) {
            part[j] += fast_silu(acc[j]) * wn;
        }
    }
    // reduce the 16 column-residues held across each 16-lane half
#pragma unroll
    for (int j = 0; j < 8; ++j) {
        float p = part[j];
        p += __shfl_xor(p, 1, 32);
        p += __shfl_xor(p, 2, 32);
        p += __shfl_xor(p, 4, 32);
        p += __shfl_xor(p, 8, 32);
        part[j] = p;
    }
    if ((lane & 15) == 0) {
        float ab2v = ab2[0];
#pragma unroll
        for (int j = 0; j < 8; ++j) {
            int node = node0 + mbase + j;
            if (node < Ntot) {
                float lg = part[j] + ab2v;
                logits[node] = lg;
                atomicMax(&m_u[batch[node]], fmap(lg));
            }
        }
    }
}

// ---------- kernel 3: exp + segment denom ----------
__global__ __launch_bounds__(256) void exp_kernel(
    const float* __restrict__ logits, const int* __restrict__ batch,
    const unsigned* __restrict__ m_u, float* __restrict__ ew,
    float* __restrict__ denom, int Ntot)
{
    int i = blockIdx.x * blockDim.x + threadIdx.x;
    if (i < Ntot) {
        int g = batch[i];
        float e = __expf(logits[i] - funmap(m_u[g]));
        ew[i] = e;
        atomicAdd(&denom[g], e);
    }
}

// ---------- kernel 4: 3-layer MLP + weighted segment-sum readout ----------
__global__ __launch_bounds__(128) void mlp_kernel(
    const float* __restrict__ X, const int* __restrict__ batch,
    const bf16_t* __restrict__ Wb1, const bf16_t* __restrict__ Wb2,
    const bf16_t* __restrict__ Wb3,
    const float* __restrict__ b1, const float* __restrict__ b2,
    const float* __restrict__ b3,
    const float* __restrict__ ew, const float* __restrict__ denom,
    float* __restrict__ out, int Ntot)
{
    __shared__ __align__(16) bf16_t bufA[4 * 16 * 256];
    __shared__ __align__(16) bf16_t bufB[4 * 16 * 256];
    __shared__ float s_b1[256], s_b2[256], s_b3[256];
    __shared__ float s_w[4][16];
    __shared__ int   s_g[4][16];

    const int wv   = threadIdx.x >> 5;
    const int lane = threadIdx.x & 31;
    const int node0 = (blockIdx.x * 4 + wv) * 16;
    bf16_t* sa = bufA + wv * 4096;
    bf16_t* sb = bufB + wv * 4096;

    for (int i = threadIdx.x; i < 256; i += 128) {
        s_b1[i] = b1[i]; s_b2[i] = b2[i]; s_b3[i] = b3[i];
    }
    if (lane < 16) {                       // per-node softmax weight + graph id
        int node = node0 + lane;
        float w; int g;
        if (node < Ntot) { g = batch[node]; w = ew[node] / denom[g]; }
        else             { g = batch[Ntot - 1]; w = 0.f; }
        s_w[wv][lane] = w; s_g[wv][lane] = g;
    }
    // stage 16x256 node-feature strip as bf16
    const float4* X4 = (const float4*)X;
    for (int i = 0; i < 32; ++i) {
        int idx4 = i * 32 + lane;
        int m = idx4 >> 6, c4 = idx4 & 63;
        int node = node0 + m;
        float4 f;
        if (node < Ntot) f = X4[(size_t)node * 64 + c4];
        else             f = make_float4(0.f, 0.f, 0.f, 0.f);
        v4bf b; b[0] = (bf16_t)f.x; b[1] = (bf16_t)f.y; b[2] = (bf16_t)f.z; b[3] = (bf16_t)f.w;
        *(v4bf*)(sa + m * 256 + c4 * 4) = b;
    }
    __syncthreads();

    layer_silu(sa, sb, Wb1, s_b1, lane);   // h1 = silu(X W1 + b1)
    __syncthreads();
    layer_silu(sb, sa, Wb2, s_b2, lane);   // h2 = silu(h1 W2 + b2)
    __syncthreads();

    // final layer: x = h2 W3 + b3, then weighted segment-sum into out
    union Frag { v16bf v; v8bf h[2]; };
    Frag A[8], B;
    const int mrow = lane & 15, khalf = (lane >> 4) * 8;
#pragma unroll
    for (int kt = 0; kt < 8; ++kt) {
        int off = mrow * 256 + kt * 32 + khalf;
        A[kt].h[0] = *(const v8bf*)(sa + off);
        A[kt].h[1] = *(const v8bf*)(sa + off + 16);
    }
    const int ncol = lane & 15, mbase = (lane >> 4) * 8;
    for (int nt = 0; nt < 16; ++nt) {
        float bv = s_b3[nt * 16 + ncol];
        v8f acc;
#pragma unroll
        for (int j = 0; j < 8; ++j) acc[j] = bv;
#pragma unroll
        for (int kt = 0; kt < 8; ++kt) {
            const bf16_t* wp = Wb3 + ((nt * 8 + kt) * 512 + lane * 16);
            B.h[0] = *(const v8bf*)(wp);
            B.h[1] = *(const v8bf*)(wp + 8);
            acc = __builtin_amdgcn_wmma_f32_16x16x32_bf16(
                false, A[kt].v, false, B.v, (short)0, acc, false, false);
        }
        // sorted segments: accumulate a run, flush on graph-id change
        float runsum = 0.f;
        int curg = s_g[wv][mbase];
#pragma unroll
        for (int j = 0; j < 8; ++j) {
            int m = mbase + j;
            int g = s_g[wv][m];
            float c = acc[j] * s_w[wv][m];
            if (g != curg) {
                atomicAdd(&out[(size_t)curg * 256 + nt * 16 + ncol], runsum);
                runsum = 0.f; curg = g;
            }
            runsum += c;
        }
        atomicAdd(&out[(size_t)curg * 256 + nt * 16 + ncol], runsum);
    }
}

// ---------- host launch ----------
extern "C" void kernel_launch(void* const* d_in, const int* in_sizes, int n_in,
                              void* d_out, int out_size, void* d_ws, size_t ws_size,
                              hipStream_t stream)
{
    const float* X    = (const float*)d_in[0];
    const int*   batch= (const int*)  d_in[1];
    const float* W1   = (const float*)d_in[3];
    const float* b1   = (const float*)d_in[4];
    const float* W2   = (const float*)d_in[5];
    const float* b2   = (const float*)d_in[6];
    const float* W3   = (const float*)d_in[7];
    const float* b3   = (const float*)d_in[8];
    const float* aW1  = (const float*)d_in[9];
    const float* ab1  = (const float*)d_in[10];
    const float* aW2  = (const float*)d_in[11];
    const float* ab2  = (const float*)d_in[12];
    const int Ntot = in_sizes[0] / 256;
    const int G    = out_size / 256;
    float* out = (float*)d_out;

    // workspace layout
    char* ws = (char*)d_ws;
    bf16_t* Wb1 = (bf16_t*)(ws + 0 * (size_t)(1 << 17));
    bf16_t* Wb2 = (bf16_t*)(ws + 1 * (size_t)(1 << 17));
    bf16_t* Wb3 = (bf16_t*)(ws + 2 * (size_t)(1 << 17));
    bf16_t* WbA = (bf16_t*)(ws + 3 * (size_t)(1 << 17));
    size_t off = 4 * (size_t)(1 << 17);
    float* logits = (float*)(ws + off); off += (((size_t)Ntot * 4) + 255) & ~(size_t)255;
    float* ewp    = (float*)(ws + off); off += (((size_t)Ntot * 4) + 255) & ~(size_t)255;
    unsigned* m_u = (unsigned*)(ws + off); off += (((size_t)G * 4) + 255) & ~(size_t)255;
    float* denom  = (float*)(ws + off);

    convert_weights<<<1024, 256, 0, stream>>>(W1, W2, W3, aW1, Wb1, Wb2, Wb3, WbA);
    init_kernel<<<(out_size + 255) / 256, 256, 0, stream>>>(out, m_u, denom, out_size, G);

    const int nblocks = (Ntot + 63) / 64;                 // 4 waves x 16 nodes
    attn_kernel<<<nblocks, 128, 0, stream>>>(X, batch, WbA, ab1, aW2, ab2,
                                             logits, m_u, Ntot);
    exp_kernel<<<(Ntot + 255) / 256, 256, 0, stream>>>(logits, batch, m_u, ewp, denom, Ntot);
    mlp_kernel<<<nblocks, 128, 0, stream>>>(X, batch, Wb1, Wb2, Wb3, b1, b2, b3,
                                            ewp, denom, out, Ntot);
}
